// GraphSAGE_45707041964681
// MI455X (gfx1250) — matrix operand order repaired and checked by reading, not compiled, not run on hardware
//
#include <hip/hip_runtime.h>
#include <hip/hip_bf16.h>

// ---------------------------------------------------------------------------
// GraphSAGE on gfx1250 (MI455X):
//   k0: zero accumulators (agg, deg, pool_acc, cnt)
//   k1: edge scatter  agg[dst]+=h[src], deg[dst]+=1   (wave-per-edge, f32 atomics)
//   k2: fused GEMM    hid = relu([h | agg/deg] @ [Wself;Wneigh] + b)  (WMMA bf16)
//       - full 256x256 B staged once in LDS (132 KB, padded, ~conflict-free)
//       - barrier-free K loop, software-pipelined B frags, prefetched A
//   k3: avg-pool per graph (sorted graph_ids -> chunked private accumulation)
//   k4: head          out = pooled @ Wpred + bpred
// ---------------------------------------------------------------------------

typedef __bf16 bf16_t;
typedef bf16_t v16bf __attribute__((ext_vector_type(16)));
typedef bf16_t v8bf  __attribute__((ext_vector_type(8)));
typedef float  v8f   __attribute__((ext_vector_type(8)));

#define BPAD 264   // LDS row stride (elements) for 256-wide B rows: 4*i mod 64 banks

__device__ __forceinline__ void atomicAddF(float* p, float v) {
    unsafeAtomicAdd(p, v);   // hardware global_atomic_add_f32
}

// --------------------------- k0: zero scratch ------------------------------
__global__ void zero_kernel(float* __restrict__ p, long long n) {
    long long i = (long long)blockIdx.x * blockDim.x + threadIdx.x;
    long long stride = (long long)gridDim.x * blockDim.x;
    for (; i < n; i += stride) p[i] = 0.0f;
}

// --------------------------- k1: edge scatter ------------------------------
// one wave32 per edge: 32 lanes x float4 = 128 features
__global__ void edge_scatter(const float* __restrict__ h,
                             const int* __restrict__ src,
                             const int* __restrict__ dst,
                             float* __restrict__ agg,
                             float* __restrict__ deg,
                             int n_edges) {
    int gtid   = blockIdx.x * blockDim.x + threadIdx.x;
    int wave   = gtid >> 5;
    int lane   = threadIdx.x & 31;
    int nwaves = (gridDim.x * blockDim.x) >> 5;
    for (int e = wave; e < n_edges; e += nwaves) {
        int s = src[e];
        int d = dst[e];
        const float4 v = *(const float4*)(h + (size_t)s * 128 + lane * 4);
        float* out = agg + (size_t)d * 128 + lane * 4;
        atomicAddF(out + 0, v.x);
        atomicAddF(out + 1, v.y);
        atomicAddF(out + 2, v.z);
        atomicAddF(out + 3, v.w);
        if (lane == 0) atomicAddF(&deg[d], 1.0f);
    }
}

// --------------------------- k2: fused WMMA GEMM ---------------------------
// A = [h | agg*invdeg]  (M x 256, bf16 on the fly)
// B = [Wself ; Wneigh]  (256 x 256) staged ONCE in LDS, transposed [n][k]
// Block: 256 threads = 8 waves; each wave owns a 16-row tile, all 256 cols.
__global__ __launch_bounds__(256)
void sage_gemm(const float* __restrict__ h,
               const float* __restrict__ agg,
               const float* __restrict__ deg,
               const float* __restrict__ Wself,
               const float* __restrict__ Wneigh,
               const float* __restrict__ bias,
               float* __restrict__ hid,
               int n_nodes) {
    // Bs[n][k] transposed, padded row stride: 256 * 264 * 2B = 132 KB
    __shared__ __attribute__((aligned(64))) bf16_t Bs[256 * BPAD];

    const int tid  = threadIdx.x;
    const int wave = tid >> 5;
    const int lane = tid & 31;
    const int nIdx = lane & 15;

    const int rowbase = blockIdx.x * 128 + wave * 16;
    const int r  = rowbase + nIdx;
    const int rc = (r < n_nodes) ? r : (n_nodes - 1);

    // A-fragment K offsets per ISA 16-bit A layout:
    //   lane<16: K in {kb..kb+7, kb+16..kb+23} with kb=0; lane>=16: kb=8
    const int kbA = (lane >> 4) * 8;
    // B-fragment matches A: K in {kb..kb+7, kb+16..kb+23}, kb = 0 / 8
    const int kbB = (lane >> 4) * 8;

    const float invd = 1.0f / fmaxf(deg[rc], 1.0f);

    const float* aBaseH = h   + (size_t)rc * 128;
    const float* aBaseN = agg + (size_t)rc * 128;

    // ---- stage ALL of B_cat = [Wself ; Wneigh] into LDS as bf16 ----
    // thread t: k rows {t>>3 + 32*kk}, cols (t&7)*32 .. +31  (coalesced reads)
    {
        const int s_n0 = (tid & 7) * 32;
        #pragma unroll
        for (int kk = 0; kk < 8; ++kk) {
            const int k = (tid >> 3) + kk * 32;
            const float* s = (k < 128 ? Wself + (size_t)k * 256
                                      : Wneigh + (size_t)(k - 128) * 256) + s_n0;
            #pragma unroll
            for (int j = 0; j < 32; j += 4) {
                float4 v = *(const float4*)(s + j);
                Bs[(s_n0 + j + 0) * BPAD + k] = (bf16_t)v.x;
                Bs[(s_n0 + j + 1) * BPAD + k] = (bf16_t)v.y;
                Bs[(s_n0 + j + 2) * BPAD + k] = (bf16_t)v.z;
                Bs[(s_n0 + j + 3) * BPAD + k] = (bf16_t)v.w;
            }
        }
    }
    __syncthreads();   // the only barrier: B resident for the whole K loop

    v8f acc[16] = {};

    // raw A window (prefetched one K-step ahead)
    float4 ar0, ar1, ar2, ar3;
    {
        const float* ap = aBaseH;           // ks = 0
        ar0 = *(const float4*)(ap + kbA + 0);
        ar1 = *(const float4*)(ap + kbA + 4);
        ar2 = *(const float4*)(ap + kbA + 16);
        ar3 = *(const float4*)(ap + kbA + 20);
    }

    #pragma unroll
    for (int ks = 0; ks < 8; ++ks) {
        // ---- convert current raw A to bf16 fragment ----
        const float scale = (ks >= 4) ? invd : 1.0f;
        float af[16] = {ar0.x, ar0.y, ar0.z, ar0.w, ar1.x, ar1.y, ar1.z, ar1.w,
                        ar2.x, ar2.y, ar2.z, ar2.w, ar3.x, ar3.y, ar3.z, ar3.w};
        v16bf aF;
        #pragma unroll
        for (int i = 0; i < 16; ++i) aF[i] = (bf16_t)(af[i] * scale);

        // ---- prefetch next A window (global latency hidden by 16 WMMAs) ----
        if (ks < 7) {
            const int k0n = (ks + 1) * 32;
            const float* ap = (k0n >= 128) ? (aBaseN + (k0n - 128)) : (aBaseH + k0n);
            ar0 = *(const float4*)(ap + kbA + 0);
            ar1 = *(const float4*)(ap + kbA + 4);
            ar2 = *(const float4*)(ap + kbA + 16);
            ar3 = *(const float4*)(ap + kbA + 20);
        }

        // ---- 16 column tiles, software-pipelined B fragment loads ----
        const bf16_t* brow = &Bs[(size_t)nIdx * BPAD + ks * 32 + kbB];
        v8bf lo = *(const v8bf*)(brow);
        v8bf hi = *(const v8bf*)(brow + 16);
        #pragma unroll
        for (int ct = 0; ct < 16; ++ct) {
            v16bf bF = __builtin_shufflevector(lo, hi, 0, 1, 2, 3, 4, 5, 6, 7,
                                               8, 9, 10, 11, 12, 13, 14, 15);
            if (ct < 15) {               // issue next frag before this WMMA
                const bf16_t* bn = brow + (size_t)(ct + 1) * 16 * BPAD;
                lo = *(const v8bf*)(bn);
                hi = *(const v8bf*)(bn + 16);
            }
            acc[ct] = __builtin_amdgcn_wmma_f32_16x16x32_bf16(
                false, aF, false, bF, (short)0, acc[ct], false, false);
        }
    }

    // ---- epilogue: bias + relu, store (64B contiguous per half-wave row) ----
    const int mBase = rowbase + ((lane >> 4) * 8);
    #pragma unroll
    for (int ct = 0; ct < 16; ++ct) {
        const int n  = ct * 16 + nIdx;
        const float bv = bias[n];
        #pragma unroll
        for (int j = 0; j < 8; ++j) {
            const int m = mBase + j;
            if (m < n_nodes) {
                float v = acc[ct][j] + bv;
                hid[(size_t)m * 256 + n] = v > 0.0f ? v : 0.0f;
            }
        }
    }
}

// --------------------------- k3: avg-pool per graph ------------------------
#define POOL_CHUNK 64
__global__ void pool_kernel(const float* __restrict__ hid,
                            const int* __restrict__ gids,
                            float* __restrict__ pacc,
                            float* __restrict__ cnt,
                            int n_nodes) {
    const int t     = threadIdx.x;            // channel 0..255
    const int start = blockIdx.x * POOL_CHUNK;
    const int end   = (start + POOL_CHUNK < n_nodes) ? start + POOL_CHUNK : n_nodes;
    float acc = 0.0f, c = 0.0f;
    int curg = -1;
    for (int i = start; i < end; ++i) {
        const int g = gids[i];
        if (g != curg) {
            if (curg >= 0) {
                atomicAddF(&pacc[curg * 256 + t], acc);
                if (t == 0) atomicAddF(&cnt[curg], c);
            }
            curg = g; acc = 0.0f; c = 0.0f;
        }
        acc += hid[(size_t)i * 256 + t];
        c   += 1.0f;
    }
    if (curg >= 0) {
        atomicAddF(&pacc[curg * 256 + t], acc);
        if (t == 0) atomicAddF(&cnt[curg], c);
    }
}

// --------------------------- k4: prediction head ---------------------------
__global__ void head_kernel(const float* __restrict__ pacc,
                            const float* __restrict__ cnt,
                            const float* __restrict__ Wpred,
                            const float* __restrict__ bpred,
                            float* __restrict__ out) {
    const int o = blockIdx.x * blockDim.x + threadIdx.x;
    if (o >= 64 * 40) return;
    const int g = o / 40, cls = o % 40;
    const float inv = 1.0f / fmaxf(cnt[g], 1.0f);
    float s = bpred[cls];
    for (int c = 0; c < 256; ++c)
        s += pacc[g * 256 + c] * inv * Wpred[c * 40 + cls];
    out[o] = s;
}

// ---------------------------------------------------------------------------
extern "C" void kernel_launch(void* const* d_in, const int* in_sizes, int n_in,
                              void* d_out, int out_size, void* d_ws, size_t ws_size,
                              hipStream_t stream) {
    const float* h      = (const float*)d_in[0];
    const float* Wself  = (const float*)d_in[1];
    const float* Wneigh = (const float*)d_in[2];
    const float* bias   = (const float*)d_in[3];
    const float* Wpred  = (const float*)d_in[4];
    const float* bpred  = (const float*)d_in[5];
    const int*   src    = (const int*)d_in[6];
    const int*   dst    = (const int*)d_in[7];
    const int*   gids   = (const int*)d_in[8];

    const int n_nodes = in_sizes[0] / 128;
    const int n_edges = in_sizes[6];

    // workspace layout (floats): [agg | deg | pool_acc | cnt | hid]
    float* ws   = (float*)d_ws;
    float* agg  = ws;
    float* deg  = agg  + (size_t)n_nodes * 128;
    float* pacc = deg  + n_nodes;
    float* cnt  = pacc + 64 * 256;
    float* hid  = cnt  + 64;

    const long long zn = (long long)n_nodes * 128 + n_nodes + 64 * 256 + 64;
    zero_kernel<<<2048, 256, 0, stream>>>(ws, zn);
    edge_scatter<<<4096, 256, 0, stream>>>(h, src, dst, agg, deg, n_edges);
    sage_gemm<<<(n_nodes + 127) / 128, 256, 0, stream>>>(h, agg, deg, Wself, Wneigh,
                                                         bias, hid, n_nodes);
    pool_kernel<<<(n_nodes + POOL_CHUNK - 1) / POOL_CHUNK, 256, 0, stream>>>(
        hid, gids, pacc, cnt, n_nodes);
    head_kernel<<<10, 256, 0, stream>>>(pacc, cnt, Wpred, bpred, (float*)d_out);
}
